// Graph_encoder_9723805958383
// MI455X (gfx1250) — compile-verified, matrix operand order checked by reading, not run on hardware
//
#include <hip/hip_runtime.h>
#include <hip/hip_bf16.h>
#include <math.h>

typedef __attribute__((ext_vector_type(2)))  float    v2f;
typedef __attribute__((ext_vector_type(8)))  float    v8f;
typedef __attribute__((ext_vector_type(16))) _Float16 v16h;

#define D_DIM 128

__device__ __forceinline__ float gelu_exact(float v) {
  // torch nn.GELU() default: 0.5*x*(1+erf(x/sqrt(2)))
  return 0.5f * v * (1.0f + erff(v * 0.70710678118654752440f));
}

// -------- degree / norm precompute --------
__global__ void degree_kernel(const int* __restrict__ ei, int E,
                              float* __restrict__ deg_out, float* __restrict__ deg_in) {
  int e = blockIdx.x * blockDim.x + threadIdx.x;
  if (e < E) {
    atomicAdd(&deg_out[ei[e]],     1.0f);  // row -> out-degree
    atomicAdd(&deg_in [ei[E + e]], 1.0f);  // col -> in-degree
  }
}

__global__ void rsqrt_kernel(float* __restrict__ a, int n) {
  int i = blockIdx.x * blockDim.x + threadIdx.x;
  if (i < n) a[i] = rsqrtf(a[i]);  // deg==0 -> inf, never gathered (matches ref)
}

// -------- fused WMMA GEMM: Y = [gelu?](  [gelu?](X)@W^T (+ [gelu?](X2)@W2^T)  + bias ) --------
// X:[M,128], W:[128,ldw] row-major (Y[n,d] = sum_k X[n,k]*W[d,k]), Y:[M,128]
template<bool GELU_IN, bool GELU_OUT, bool DUAL>
__global__ __launch_bounds__(256) void gemm_wmma(
    const float* __restrict__ X, const float* __restrict__ X2,
    const float* __restrict__ W, const float* __restrict__ W2, int ldw,
    const float* __restrict__ bias, float* __restrict__ Y, int M) {
  const int lane   = threadIdx.x & 31;
  const int wave   = threadIdx.x >> 5;
  const int mr     = lane & 15;   // A: row M;  B: col N
  const int kh     = lane >> 4;   // K half-select
  const int m_base = blockIdx.x * 16;
  const int n_base = wave * 16;   // 8 waves cover the 128 output dims

  int mrow = m_base + mr; if (mrow >= M) mrow = M - 1;  // clamp loads; stores guarded below
  const float* xrow  = X + (size_t)mrow * D_DIM;
  const float* x2row = DUAL ? (X2 + (size_t)mrow * D_DIM) : nullptr;
  const float* wrow  = W  + (size_t)(n_base + mr) * ldw;
  const float* w2row = DUAL ? (W2 + (size_t)(n_base + mr) * ldw) : nullptr;

  // hoist bias: same output column n for all 8 C rows of this lane
  const float bv = bias[n_base + mr];

  v8f c = {};
#if __has_builtin(__builtin_amdgcn_wmma_f32_16x16x4_f32)
  // fp32 tensor path: V_WMMA_F32_16X16X4_F32, K advances 4/instr.
  // A 16x4 layout: lanes 0-15 hold K={0,1}, lanes 16-31 K={2,3}; B mirrored on N.
  #pragma unroll 4
  for (int k0 = 0; k0 < D_DIM; k0 += 4) {
    const int ka = k0 + 2 * kh;
    v2f a, b;
    float ax = xrow[ka], ay = xrow[ka + 1];
    if (GELU_IN) { ax = gelu_exact(ax); ay = gelu_exact(ay); }
    a.x = ax; a.y = ay;
    b.x = wrow[ka]; b.y = wrow[ka + 1];
    c = __builtin_amdgcn_wmma_f32_16x16x4_f32(false, a, false, b, (short)0, c, false, false);
    if (DUAL) {
      float cx = x2row[ka], cy = x2row[ka + 1];
      if (GELU_IN) { cx = gelu_exact(cx); cy = gelu_exact(cy); }
      v2f a2, b2;
      a2.x = cx; a2.y = cy;
      b2.x = w2row[ka]; b2.y = w2row[ka + 1];
      c = __builtin_amdgcn_wmma_f32_16x16x4_f32(false, a2, false, b2, (short)0, c, false, false);
    }
  }
#else
  // fallback: codegen-confirmed f16 WMMA (fp32->fp16 inputs, fp32 accumulate)
  for (int k0 = 0; k0 < D_DIM; k0 += 32) {
    v16h a, b;
    #pragma unroll
    for (int j = 0; j < 8; ++j) {
      const int kb = k0 + (j < 4 ? 2 * j : 16 + 2 * (j - 4)) + (kh ? 8 : 0);
      float ax = xrow[kb], ay = xrow[kb + 1];
      if (GELU_IN) { ax = gelu_exact(ax); ay = gelu_exact(ay); }
      a[2 * j] = (_Float16)ax;        a[2 * j + 1] = (_Float16)ay;
      b[2 * j] = (_Float16)wrow[kb];  b[2 * j + 1] = (_Float16)wrow[kb + 1];
    }
    c = __builtin_amdgcn_wmma_f32_16x16x32_f16(false, a, false, b, (short)0, c, false, false);
    if (DUAL) {
      v16h a2, b2;
      #pragma unroll
      for (int j = 0; j < 8; ++j) {
        const int kb = k0 + (j < 4 ? 2 * j : 16 + 2 * (j - 4)) + (kh ? 8 : 0);
        float cx = x2row[kb], cy = x2row[kb + 1];
        if (GELU_IN) { cx = gelu_exact(cx); cy = gelu_exact(cy); }
        a2[2 * j] = (_Float16)cx;        a2[2 * j + 1] = (_Float16)cy;
        b2[2 * j] = (_Float16)w2row[kb]; b2[2 * j + 1] = (_Float16)w2row[kb + 1];
      }
      c = __builtin_amdgcn_wmma_f32_16x16x32_f16(false, a2, false, b2, (short)0, c, false, false);
    }
  }
#endif
  // C/D layout: VGPR r -> M = m_base + r + 8*kh, N = n_base + (lane&15)
  float* ybase = Y + (size_t)(m_base + 8 * kh) * D_DIM + (n_base + mr);
  if (m_base + 16 <= M) {
    // full tile: branch-free epilogue (always taken here: M % 16 == 0)
    #pragma unroll
    for (int r = 0; r < 8; ++r) {
      float v = c[r] + bv;
      if (GELU_OUT) v = gelu_exact(v);
      ybase[(size_t)r * D_DIM] = v;
    }
  } else {
    #pragma unroll
    for (int r = 0; r < 8; ++r) {
      const int m = m_base + r + 8 * kh;
      if (m < M) {
        float v = c[r] + bv;
        if (GELU_OUT) v = gelu_exact(v);
        ybase[(size_t)r * D_DIM] = v;
      }
    }
  }
}

// -------- fused bidirectional edge scatter --------
// one wave per edge; norm(e) = rs_out[row]*rs_in[col] is shared by both directions.
// agg_in[col]  += norm * h1[row]   (source_to_target)
// agg_out[row] += norm * h2[col]   (target_to_source)
__global__ __launch_bounds__(256) void scatter_kernel(
    const int* __restrict__ ei, int E,
    const float* __restrict__ rs_out, const float* __restrict__ rs_in,
    const float* __restrict__ h1, const float* __restrict__ h2,
    float* __restrict__ agg_in, float* __restrict__ agg_out) {
  const int e = blockIdx.x * 8 + (threadIdx.x >> 5);
  if (e >= E) return;
  const int lane = threadIdx.x & 31;
  const int r  = ei[e];
  const int cc = ei[E + e];
  const float w = rs_out[r] * rs_in[cc];
  const int d = lane * 4;
  const float4 v1 = *(const float4*)(h1 + (size_t)r  * D_DIM + d);
  const float4 v2 = *(const float4*)(h2 + (size_t)cc * D_DIM + d);
  float* pin  = agg_in  + (size_t)cc * D_DIM + d;
  float* pout = agg_out + (size_t)r  * D_DIM + d;
  atomicAdd(pin + 0, w * v1.x); atomicAdd(pin + 1, w * v1.y);
  atomicAdd(pin + 2, w * v1.z); atomicAdd(pin + 3, w * v1.w);
  atomicAdd(pout + 0, w * v2.x); atomicAdd(pout + 1, w * v2.y);
  atomicAdd(pout + 2, w * v2.z); atomicAdd(pout + 3, w * v2.w);
}

extern "C" void kernel_launch(void* const* d_in, const int* in_sizes, int n_in,
                              void* d_out, int out_size, void* d_ws, size_t ws_size,
                              hipStream_t stream) {
  const float* x0 = (const float*)d_in[0];
  const int*   ei = (const int*)d_in[1];
  const int N = in_sizes[0] / D_DIM;
  const int E = in_sizes[1] / 2;
  const size_t ND = (size_t)N * D_DIM;
  const int mg = (N + 15) / 16;

  float* ws     = (float*)d_ws;
  float* rs_out = ws;                 // [N]
  float* rs_in  = ws + N;             // [N]
  float* A  = ws + 2 * (size_t)N;     // [N,128]
  float* B  = A + ND;                 // [N,128]
  float* C  = B + ND;                 // [N,128]
  float* Db = C + ND;                 // [N,128]

  // symmetric-norm factors
  hipMemsetAsync(rs_out, 0, 2 * (size_t)N * sizeof(float), stream);
  degree_kernel<<<(E + 255) / 256, 256, 0, stream>>>(ei, E, rs_out, rs_in);
  rsqrt_kernel<<<(2 * N + 255) / 256, 256, 0, stream>>>(rs_out, 2 * N);

  const int sgrid = (E + 7) / 8;

  // ---------------- layer 0 ----------------
  {
    const float *Wi = (const float*)d_in[2], *bi = (const float*)d_in[3];
    const float *Wo = (const float*)d_in[4], *bo = (const float*)d_in[5];
    const float *W1 = (const float*)d_in[6], *b1 = (const float*)d_in[7];
    const float *W2 = (const float*)d_in[8], *b2 = (const float*)d_in[9];

    gemm_wmma<false,false,false><<<mg,256,0,stream>>>(x0, nullptr, Wi, nullptr, D_DIM, bi, A, N); // h1
    gemm_wmma<false,false,false><<<mg,256,0,stream>>>(x0, nullptr, Wo, nullptr, D_DIM, bo, B, N); // h2
    hipMemsetAsync(C,  0, ND * sizeof(float), stream);   // agg_in
    hipMemsetAsync(Db, 0, ND * sizeof(float), stream);   // agg_out
    scatter_kernel<<<sgrid,256,0,stream>>>(ei, E, rs_out, rs_in, A, B, C, Db);
    // t = gelu( gelu(agg_in)@W1a^T + gelu(agg_out)@W1b^T + b1 )  (W1 is [128,256])
    gemm_wmma<true,true,true><<<mg,256,0,stream>>>(C, Db, W1, W1 + D_DIM, 2 * D_DIM, b1, A, N);
    gemm_wmma<false,false,false><<<mg,256,0,stream>>>(A, nullptr, W2, nullptr, D_DIM, b2, B, N);  // x1 -> B
  }

  // ---------------- layer 1 (input = B) ----------------
  {
    const float *Wi = (const float*)d_in[10], *bi = (const float*)d_in[11];
    const float *Wo = (const float*)d_in[12], *bo = (const float*)d_in[13];
    const float *W1 = (const float*)d_in[14], *b1 = (const float*)d_in[15];
    const float *W2 = (const float*)d_in[16], *b2 = (const float*)d_in[17];

    gemm_wmma<false,false,false><<<mg,256,0,stream>>>(B, nullptr, Wi, nullptr, D_DIM, bi, A, N);  // h1
    gemm_wmma<false,false,false><<<mg,256,0,stream>>>(B, nullptr, Wo, nullptr, D_DIM, bo, C, N);  // h2
    hipMemsetAsync(Db, 0, ND * sizeof(float), stream);   // agg_in
    hipMemsetAsync(B,  0, ND * sizeof(float), stream);   // agg_out (x1 no longer needed)
    scatter_kernel<<<sgrid,256,0,stream>>>(ei, E, rs_out, rs_in, A, C, Db, B);
    gemm_wmma<true,true,true><<<mg,256,0,stream>>>(Db, B, W1, W1 + D_DIM, 2 * D_DIM, b1, A, N);
    gemm_wmma<false,false,false><<<mg,256,0,stream>>>(A, nullptr, W2, nullptr, D_DIM, b2,
                                                      (float*)d_out, N);
  }
}
